// MultiHeadAttention_82884278878701
// MI455X (gfx1250) — matrix operand order repaired and checked by reading, not compiled, not run on hardware
//
#include <hip/hip_runtime.h>
#include <hip/hip_bf16.h>

#define B_  4
#define S_  2048
#define H_  1024
#define NH_ 16
#define HS_ 64

typedef unsigned int   u32;
typedef unsigned short u16;
typedef __attribute__((ext_vector_type(16))) __bf16 bf16x16;
typedef __attribute__((ext_vector_type(8)))  float  f32x8;

union FragU { bf16x16 v; uint4 q[2]; };

// ---- CDNA5 async global->LDS path (guarded; falls back to sync copy) ------
#if defined(__has_builtin)
#  if __has_builtin(__builtin_amdgcn_global_load_async_to_lds_b128)
#    define USE_ASYNC_LDS 1
#  endif
#endif
#ifndef USE_ASYNC_LDS
#  define USE_ASYNC_LDS 0
#endif

typedef int i32x4v __attribute__((vector_size(16)));
typedef __attribute__((address_space(1))) i32x4v gvec4i;   // global
typedef __attribute__((address_space(3))) i32x4v lvec4i;   // LDS

__device__ __forceinline__ void async_ld_b128(void* lds_dst, const void* gsrc) {
#if USE_ASYNC_LDS
  __builtin_amdgcn_global_load_async_to_lds_b128(
      (gvec4i*)gsrc, (lvec4i*)lds_dst, 0, 0);
#else
  *(uint4*)lds_dst = *(const uint4*)gsrc;
#endif
}

__device__ __forceinline__ void wait_async() {
#if USE_ASYNC_LDS
#  if __has_builtin(__builtin_amdgcn_s_wait_asynccnt)
  __builtin_amdgcn_s_wait_asynccnt(0);
#  else
  asm volatile("s_wait_asynccnt 0" ::: "memory");
#  endif
#endif
}

__device__ __forceinline__ u16 f2bf(float f) {
  u32 u = __builtin_bit_cast(u32, f);
  u32 r = u + 0x7FFFu + ((u >> 16) & 1u);   // round-to-nearest-even
  return (u16)(r >> 16);
}

__device__ __forceinline__ void lds_fence_wave() {
  asm volatile("s_wait_dscnt 0" ::: "memory");
}

// A-matrix fragment (16-bit 16x32, ISA 7.12.2): lane row = `row`,
// K chunks {kb+8*half .. +7} and {kb+16+8*half .. +7}
__device__ __forceinline__ bf16x16 frag_a(const u16* lds, int row, int pitch, int kb, int half) {
  FragU f;
  const u16* p = lds + row * pitch + kb + 8 * half;
  f.q[0] = *(const uint4*)(p);
  f.q[1] = *(const uint4*)(p + 16);
  return f.v;
}
// B-matrix fragment (16-bit 32x16): lane col = `col`,
// K chunks {kb+8*half .. +7} and {kb+16+8*half .. +7}
__device__ __forceinline__ bf16x16 frag_b(const u16* lds, int col, int pitch, int kb, int half) {
  FragU f;
  const u16* p = lds + col * pitch + kb + 8 * half;
  f.q[0] = *(const uint4*)(p);
  f.q[1] = *(const uint4*)(p + 16);
  return f.v;
}

__device__ __forceinline__ f32x8 wmma_bf16(bf16x16 a, bf16x16 b, f32x8 c) {
  return __builtin_amdgcn_wmma_f32_16x16x32_bf16(false, a, false, b, (short)0, c, false, false);
}

// ---------------------------------------------------------------------------
// Kernel 1: fused Q/K/V projection GEMM.  C[m][n] = sum_k x[m][k]*W[n][k] + b[n]
// grid = (B*S/64, 3H/64), block = 128 (4 waves).  Q scaled by 1/sqrt(HS).
// Q,K stored (b,h,s,d) bf16; V stored transposed (b,h,d,s) bf16.
// ---------------------------------------------------------------------------
__global__ __launch_bounds__(128) void qkv_kernel(
    const float* __restrict__ x,
    const float* __restrict__ Wq, const float* __restrict__ bq,
    const float* __restrict__ Wk, const float* __restrict__ bk,
    const float* __restrict__ Wv, const float* __restrict__ bv,
    u16* __restrict__ Qb, u16* __restrict__ Kb, u16* __restrict__ Vt) {
  __shared__ alignas(16) u16 As[64 * 64];
  __shared__ alignas(16) u16 Bs[64 * 64];

  const int mBase = blockIdx.x * 64;
  const int nBase = blockIdx.y * 64;          // [0, 3H)
  const int which = nBase / H_;               // 0=Q 1=K 2=V (H%64==0 -> uniform)
  const int nb    = nBase % H_;
  const float* W    = (which == 0) ? Wq : (which == 1) ? Wk : Wv;
  const float* bias = (which == 0) ? bq : (which == 1) ? bk : bv;

  const int tid  = threadIdx.x;
  const int wave = tid >> 5, lane = tid & 31;
  const int half = lane >> 4, lr = lane & 15;

  f32x8 acc[4] = {};

  for (int k0 = 0; k0 < H_; k0 += 64) {
    __syncthreads();
    // stage A (x tile) and B (W tile), f32 -> bf16, 64x64 each
#pragma unroll
    for (int i = 0; i < 16; ++i) {
      int idx = tid + i * 128;                // pair index, 2048 pairs per tile
      int r = idx >> 5, c2 = idx & 31;
      float2 xv = *(const float2*)(x + (size_t)(mBase + r) * H_ + k0 + 2 * c2);
      *(u32*)(As + r * 64 + 2 * c2) = (u32)f2bf(xv.x) | ((u32)f2bf(xv.y) << 16);
      float2 wv = *(const float2*)(W + (size_t)(nb + r) * H_ + k0 + 2 * c2);
      *(u32*)(Bs + r * 64 + 2 * c2) = (u32)f2bf(wv.x) | ((u32)f2bf(wv.y) << 16);
    }
    __syncthreads();
#pragma unroll
    for (int kb = 0; kb < 64; kb += 32) {
      // load all fragments first, then back-to-back WMMAs
      bf16x16 a = frag_a(As, 16 * wave + lr, 64, kb, half);
      bf16x16 bb[4];
#pragma unroll
      for (int nt = 0; nt < 4; ++nt) bb[nt] = frag_b(Bs, 16 * nt + lr, 64, kb, half);
#pragma unroll
      for (int nt = 0; nt < 4; ++nt) acc[nt] = wmma_bf16(a, bb[nt], acc[nt]);
    }
  }

  // epilogue: D layout lane->(N=lr, M=v+8*half)
#pragma unroll
  for (int nt = 0; nt < 4; ++nt) {
    const int   ncol = nb + 16 * nt + lr;     // within H
    const int   h = ncol >> 6, d = ncol & 63;
    const float bval = bias[ncol];
    if (which == 0) {
#pragma unroll
      for (int v = 0; v < 8; ++v) {
        int m = mBase + 16 * wave + v + 8 * half;
        int bb = m >> 11, s = m & (S_ - 1);
        Qb[(((size_t)(bb * NH_ + h)) * S_ + s) * HS_ + d] =
            f2bf((acc[nt][v] + bval) * 0.125f);          // fold 1/sqrt(64)
      }
    } else if (which == 1) {
#pragma unroll
      for (int v = 0; v < 8; ++v) {
        int m = mBase + 16 * wave + v + 8 * half;
        int bb = m >> 11, s = m & (S_ - 1);
        Kb[(((size_t)(bb * NH_ + h)) * S_ + s) * HS_ + d] = f2bf(acc[nt][v] + bval);
      }
    } else {
      // transposed V store: 8 consecutive s -> one 16B store
      u32 pk[4];
#pragma unroll
      for (int i = 0; i < 4; ++i)
        pk[i] = (u32)f2bf(acc[nt][2 * i] + bval) |
                ((u32)f2bf(acc[nt][2 * i + 1] + bval) << 16);
      int m0 = mBase + 16 * wave + 8 * half;
      int bb = m0 >> 11, s0 = m0 & (S_ - 1);
      *(uint4*)(Vt + (((size_t)(bb * NH_ + h)) * HS_ + d) * S_ + s0) = *(uint4*)pk;
    }
  }
}

// ---------------------------------------------------------------------------
// Kernel 2: flash attention.  grid = (S/64, B*NH), block = 128 (4 waves).
// Each wave owns 16 query rows.  K/V tiles double-buffered in LDS and
// streamed with GLOBAL_LOAD_ASYNC_TO_LDS (ASYNCcnt) to overlap with WMMA.
// ---------------------------------------------------------------------------
__global__ __launch_bounds__(128) void attn_kernel(
    const u16* __restrict__ Qb, const u16* __restrict__ Kb,
    const u16* __restrict__ Vt, const int* __restrict__ mask,
    u16* __restrict__ Ctx) {
  __shared__ alignas(16) u16   Qs[64 * 64];
  __shared__ alignas(16) u16   Ks[2][64 * 64];
  __shared__ alignas(16) u16   Vs[2][64 * 64];     // [d][s_local]
  __shared__ alignas(16) float Ps[4][16 * 64];     // per-wave f32 scores
  __shared__ alignas(16) u16   Pb[4][16 * 64];     // per-wave bf16 probs
  __shared__ float rowScale[4][16];
  __shared__ float rowSum[4][16];
  __shared__ int   smask[2][64];

  const int qBase = blockIdx.x * 64;
  const int bh    = blockIdx.y;
  const int bidx  = bh >> 4, h = bh & 15;
  const u16* Qg = Qb + (size_t)bh * S_ * HS_;
  const u16* Kg = Kb + (size_t)bh * S_ * HS_;
  const u16* Vg = Vt + (size_t)bh * HS_ * S_;
  const int* mg = mask + bidx * S_;

  const int tid  = threadIdx.x;
  const int wave = tid >> 5, lane = tid & 31;
  const int half = lane >> 4, lr = lane & 15;

  // prologue: async-stage Q tile and first K/V tiles into buffer 0
#pragma unroll
  for (int i = 0; i < 4; ++i) {
    int off = (tid + i * 128) * 8;
    async_ld_b128(Qs + off, Qg + (size_t)qBase * HS_ + off);
    async_ld_b128(Ks[0] + off, Kg + off);
  }
#pragma unroll
  for (int i = 0; i < 4; ++i) {
    int idx = tid + i * 128;
    int r = idx >> 3, c = (idx & 7) * 8;
    async_ld_b128(Vs[0] + r * 64 + c, Vg + (size_t)r * S_ + c);
  }
  if (tid < 64) smask[0][tid] = mg[tid];

  f32x8 acc[4] = {};
  float m_run = -__builtin_inff();
  float l_run = 0.f;

  for (int kt = 0; kt < S_; kt += 64) {
    const int cur = (kt >> 6) & 1, nxt = cur ^ 1;
    wait_async();                     // buffer `cur` (and Q) resident in LDS
    __syncthreads();                  // everyone done writing cur / reading nxt

    if (kt + 64 < S_) {               // kick off next tile into `nxt`
#pragma unroll
      for (int i = 0; i < 4; ++i) {
        int off = (tid + i * 128) * 8;
        async_ld_b128(Ks[nxt] + off, Kg + (size_t)(kt + 64) * HS_ + off);
      }
#pragma unroll
      for (int i = 0; i < 4; ++i) {
        int idx = tid + i * 128;
        int r = idx >> 3, c = (idx & 7) * 8;
        async_ld_b128(Vs[nxt] + r * 64 + c, Vg + (size_t)r * S_ + (kt + 64) + c);
      }
      if (tid < 64) smask[nxt][tid] = mg[kt + 64 + tid];
    }
    if (kt + 128 < S_)                // CDNA5 global_prefetch_b8, one tile ahead
      __builtin_prefetch(Kg + (size_t)(kt + 128) * HS_, 0, 1);

    // scores = (Q/8) K^T   (16x64 per wave)
    f32x8 sc[4] = {};
#pragma unroll
    for (int kb = 0; kb < 64; kb += 32) {
      bf16x16 a = frag_a(Qs, 16 * wave + lr, 64, kb, half);
      bf16x16 bb[4];
#pragma unroll
      for (int nt = 0; nt < 4; ++nt) bb[nt] = frag_b(Ks[cur], 16 * nt + lr, 64, kb, half);
#pragma unroll
      for (int nt = 0; nt < 4; ++nt) sc[nt] = wmma_bf16(a, bb[nt], sc[nt]);
    }
#pragma unroll
    for (int nt = 0; nt < 4; ++nt)
#pragma unroll
      for (int v = 0; v < 8; ++v)
        Ps[wave][(v + 8 * half) * 64 + 16 * nt + lr] = sc[nt][v];
    lds_fence_wave();

    // online softmax: lane handles row lr, columns [32*half, 32*half+32)
    const float* prow = &Ps[wave][lr * 64 + 32 * half];
    const int    c0   = 32 * half;
    float tmax = -__builtin_inff();
#pragma unroll
    for (int i = 0; i < 32; ++i) {
      float s = (smask[cur][c0 + i] != 0) ? prow[i] : -__builtin_inff();
      tmax = fmaxf(tmax, s);
    }
    tmax = fmaxf(tmax, __shfl_xor(tmax, 16, 32));
    float mnew  = fmaxf(m_run, tmax);
    float alpha = (mnew == -__builtin_inff()) ? 1.0f : __expf(m_run - mnew);
    float psum  = 0.f;
#pragma unroll
    for (int i = 0; i < 32; ++i) {
      float s = (smask[cur][c0 + i] != 0) ? prow[i] : -__builtin_inff();
      float p = (s == -__builtin_inff()) ? 0.f : __expf(s - mnew);
      psum += p;
      Pb[wave][lr * 64 + c0 + i] = f2bf(p);
    }
    psum += __shfl_xor(psum, 16, 32);
    l_run = l_run * alpha + psum;
    m_run = mnew;
    if (half == 0) rowScale[wave][lr] = alpha;
    lds_fence_wave();

    // rescale accumulator rows (M = v + 8*half)
#pragma unroll
    for (int v = 0; v < 8; ++v) {
      float sc8 = rowScale[wave][v + 8 * half];
#pragma unroll
      for (int nt = 0; nt < 4; ++nt) acc[nt][v] *= sc8;
    }

    // out += P @ V
#pragma unroll
    for (int kb = 0; kb < 64; kb += 32) {
      bf16x16 a = frag_a(Pb[wave], lr, 64, kb, half);
      bf16x16 bb[4];
#pragma unroll
      for (int nt = 0; nt < 4; ++nt) bb[nt] = frag_b(Vs[cur], 16 * nt + lr, 64, kb, half);
#pragma unroll
      for (int nt = 0; nt < 4; ++nt) acc[nt] = wmma_bf16(a, bb[nt], acc[nt]);
    }
  }

  if (half == 0) rowSum[wave][lr] = l_run;
  lds_fence_wave();
#pragma unroll
  for (int v = 0; v < 8; ++v) {
    float inv = 1.0f / rowSum[wave][v + 8 * half];
    int   sg  = qBase + 16 * wave + v + 8 * half;
#pragma unroll
    for (int nt = 0; nt < 4; ++nt)
      Ctx[((size_t)(bidx * S_ + sg)) * H_ + h * HS_ + 16 * nt + lr] =
          f2bf(acc[nt][v] * inv);
  }
}

// ---------------------------------------------------------------------------
// Kernel 3: output projection.  out[m][n] = sum_k ctx[m][k]*Wo[n][k] + bo[n]
// grid = (B*S/64, H/64), block = 128.
// ---------------------------------------------------------------------------
__global__ __launch_bounds__(128) void oproj_kernel(
    const u16* __restrict__ Ctx, const float* __restrict__ Wo,
    const float* __restrict__ bo, float* __restrict__ out) {
  __shared__ alignas(16) u16 As[64 * 64];
  __shared__ alignas(16) u16 Bs[64 * 64];
  const int mBase = blockIdx.x * 64;
  const int nBase = blockIdx.y * 64;
  const int tid  = threadIdx.x;
  const int wave = tid >> 5, lane = tid & 31;
  const int half = lane >> 4, lr = lane & 15;

  f32x8 acc[4] = {};
  for (int k0 = 0; k0 < H_; k0 += 64) {
    __syncthreads();
#pragma unroll
    for (int i = 0; i < 4; ++i) {            // A: bf16 copy
      int idx = tid + i * 128;
      int r = idx >> 3, c = (idx & 7) * 8;
      *(uint4*)(As + r * 64 + c) = *(const uint4*)(Ctx + (size_t)(mBase + r) * H_ + k0 + c);
    }
#pragma unroll
    for (int i = 0; i < 16; ++i) {           // B: f32 -> bf16
      int idx = tid + i * 128;
      int r = idx >> 5, c2 = idx & 31;
      float2 wv = *(const float2*)(Wo + (size_t)(nBase + r) * H_ + k0 + 2 * c2);
      *(u32*)(Bs + r * 64 + 2 * c2) = (u32)f2bf(wv.x) | ((u32)f2bf(wv.y) << 16);
    }
    __syncthreads();
#pragma unroll
    for (int kb = 0; kb < 64; kb += 32) {
      bf16x16 a = frag_a(As, 16 * wave + lr, 64, kb, half);
      bf16x16 bb[4];
#pragma unroll
      for (int nt = 0; nt < 4; ++nt) bb[nt] = frag_b(Bs, 16 * nt + lr, 64, kb, half);
#pragma unroll
      for (int nt = 0; nt < 4; ++nt) acc[nt] = wmma_bf16(a, bb[nt], acc[nt]);
    }
  }
#pragma unroll
  for (int nt = 0; nt < 4; ++nt) {
    int   n = nBase + 16 * nt + lr;
    float bval = bo[n];
#pragma unroll
    for (int v = 0; v < 8; ++v) {
      int m = mBase + 16 * wave + v + 8 * half;
      out[(size_t)m * H_ + n] = acc[nt][v] + bval;
    }
  }
}

// ---------------------------------------------------------------------------
extern "C" void kernel_launch(void* const* d_in, const int* in_sizes, int n_in,
                              void* d_out, int out_size, void* d_ws, size_t ws_size,
                              hipStream_t stream) {
  (void)in_sizes; (void)n_in; (void)out_size; (void)ws_size;
  const float* x  = (const float*)d_in[0];
  const int*  msk = (const int*)d_in[1];
  const float* Wq = (const float*)d_in[2];
  const float* bq = (const float*)d_in[3];
  const float* Wk = (const float*)d_in[4];
  const float* bk = (const float*)d_in[5];
  const float* Wv = (const float*)d_in[6];
  const float* bv = (const float*)d_in[7];
  const float* Wo = (const float*)d_in[8];
  const float* bo = (const float*)d_in[9];
  float* out = (float*)d_out;

  const size_t elems = (size_t)B_ * NH_ * S_ * HS_;   // 8M elements per buffer
  u16* Qb  = (u16*)d_ws;
  u16* Kb  = Qb + elems;
  u16* Vt  = Kb + elems;
  u16* Ctx = Vt + elems;                               // total 64 MB of ws

  qkv_kernel<<<dim3((B_ * S_) / 64, (3 * H_) / 64), 128, 0, stream>>>(
      x, Wq, bq, Wk, bk, Wv, bv, Qb, Kb, Vt);
  attn_kernel<<<dim3(S_ / 64, B_ * NH_), 128, 0, stream>>>(Qb, Kb, Vt, msk, Ctx);
  oproj_kernel<<<dim3((B_ * S_) / 64, H_ / 64), 128, 0, stream>>>(Ctx, Wo, bo, out);
}
